// SEM_25177098289657
// MI455X (gfx1250) — compile-verified, matrix operand order checked by reading, not run on hardware
//
#include <hip/hip_runtime.h>
#include <hip/hip_bf16.h>
#include <math.h>

typedef __attribute__((ext_vector_type(16))) _Float16 v16h;
typedef __attribute__((ext_vector_type(8)))  float    v8f;

#define TPB 256

// =====================================================================
// WMMA implicit-GEMM convolution (NCHW fp32 in/out, f16 WMMA compute)
// M = Cout, N = Nb*Ho*Wo pixels, K = Cin*ks*ks
// Block = 256 threads = 8 waves arranged MW x NWW; each wave computes a
// 16 x 32 output tile (one A fragment, two B fragments, two WMMAs/chunk).
// A (weights) staged via gfx1250 async global->LDS DMA (ASYNCcnt).
// Supports fused 2-tensor channel concat on the input (x: Cin1 ch, x2: rest).
// =====================================================================
template<int MW, int NWW>
__global__ __launch_bounds__(TPB)
void conv_wmma_kernel(const float* __restrict__ x, const float* __restrict__ x2,
                      int Cin1, const float* __restrict__ w,
                      const float* __restrict__ bias, float* __restrict__ y,
                      int Nb, int Cin, int H, int W, int Cout, int Ho, int Wo,
                      int ks, int pad, int dil, int stride, int relu)
{
    constexpr int MT  = MW * 16;         // M tile (rows of weights)
    constexpr int NT  = NWW * 32;        // N tile (pixels)
    constexpr int NTS = (NWW == 2) ? 6 : ((NWW == 4) ? 7 : 8);  // log2(NT)
    __shared__ float    Asf[MT][32];     // f32 weights chunk (async DMA target)
    __shared__ _Float16 Bs[32][NT + 8];  // f16 im2col patches

    const int K    = Cin * ks * ks;
    const int KC   = (K + 31) / 32;
    const int HoWo = Ho * Wo;
    const int Npix = Nb * HoWo;

    const int nbase = blockIdx.x * NT;
    const int mbase = blockIdx.y * MT;

    const int tid  = threadIdx.x;
    const int lane = tid & 31;
    const int wid  = tid >> 5;
    const int wm   = wid % MW;
    const int wn   = wid / MW;

    const long wClampMax = (long)Cout * K - 4;  // clamp async reads into buffer

    v8f acc0 = {};
    v8f acc1 = {};

    for (int kc = 0; kc < KC; ++kc) {
        const int k0 = kc * 32;

        // ---- stage A (weights) via async global->LDS B128 DMA ----
        for (int t2 = tid; t2 < MT * 8; t2 += TPB) {
            int ml = t2 >> 3, q = t2 & 7;            // row, 4-float group
            long koff = (long)(mbase + ml) * K + k0 + q * 4;
            if (koff > wClampMax) koff = wClampMax;  // finite (masked) data
            if (koff < 0) koff = 0;
            unsigned ldsa = (unsigned)(uintptr_t)(void*)&Asf[ml][q * 4];
            const float* gsrc = w + koff;
            asm volatile("global_load_async_to_lds_b128 %0, %1, off"
                         :: "v"(ldsa), "v"(gsrc) : "memory");
        }

        // ---- stage B (im2col, pixel index in low bits -> coalesced) ----
        for (int idx = tid; idx < 32 * NT; idx += TPB) {
            int kl = idx >> NTS;
            int j  = idx & (NT - 1);
            int kg = k0 + kl;
            int gp = nbase + j;
            float v = 0.f;
            if (kg < K && gp < Npix) {
                int n   = gp / HoWo;
                int rem = gp - n * HoWo;
                int oy  = rem / Wo;
                int ox  = rem - oy * Wo;
                int kk2 = ks * ks;
                int ci  = kg / kk2;
                int kr  = kg - ci * kk2;
                int kh  = kr / ks;
                int kw  = kr - kh * ks;
                int iy  = oy * stride - pad + kh * dil;
                int ix  = ox * stride - pad + kw * dil;
                if (iy >= 0 && iy < H && ix >= 0 && ix < W) {
                    if (ci < Cin1)
                        v = x[(((size_t)n * Cin1 + ci) * H + iy) * W + ix];
                    else
                        v = x2[(((size_t)n * (Cin - Cin1) + (ci - Cin1)) * H + iy) * W + ix];
                }
            }
            Bs[kl][j] = (_Float16)v;
        }

        asm volatile("s_wait_asynccnt 0" ::: "memory");
        __syncthreads();

        // ---- build fragments per ISA 16-bit A/B lane layout ----
        v16h a, b0, b1;
        #pragma unroll
        for (int v = 0; v < 16; ++v) {
            int k = (v & 7) + ((v >> 3) << 4) + ((lane >> 4) << 3);
            a[v]  = (_Float16)Asf[wm * 16 + (lane & 15)][k];
            b0[v] = Bs[k][wn * 32 + (lane & 15)];
            b1[v] = Bs[k][wn * 32 + 16 + (lane & 15)];
        }
        acc0 = __builtin_amdgcn_wmma_f32_16x16x32_f16(false, a, false, b0,
                                                      (short)0, acc0, false, false);
        acc1 = __builtin_amdgcn_wmma_f32_16x16x32_f16(false, a, false, b1,
                                                      (short)0, acc1, false, false);
        __syncthreads();
    }

    // ---- store D per 16x16 f32 C/D layout (two N-subtiles) ----
    #pragma unroll
    for (int r = 0; r < 8; ++r) {
        int m = mbase + wm * 16 + r + ((lane >> 4) << 3);
        if (m >= Cout) continue;
        float bv = bias ? bias[m] : 0.f;
        #pragma unroll
        for (int half = 0; half < 2; ++half) {
            int gp = nbase + wn * 32 + half * 16 + (lane & 15);
            if (gp < Npix) {
                int n   = gp / HoWo;
                int rem = gp - n * HoWo;
                float v = (half ? acc1[r] : acc0[r]) + bv;
                if (relu) v = fmaxf(v, 0.f);
                y[((size_t)n * Cout + m) * HoWo + rem] = v;
            }
        }
    }
}

// =====================================================================
// BatchNorm (no affine): per-channel mean/var over (N,H,W), then normalize
// =====================================================================
__global__ __launch_bounds__(TPB)
void bn_stats_kernel(const float* __restrict__ x, float* __restrict__ stats,
                     int Nb, int C, int HW)
{
    int c = blockIdx.x;
    int per = Nb * HW;
    float s = 0.f, s2 = 0.f;
    for (int i = threadIdx.x; i < per; i += TPB) {
        int n = i / HW, r = i - n * HW;
        float v = x[((size_t)n * C + c) * HW + r];
        s += v; s2 += v * v;
    }
    __shared__ float ss[TPB], sq[TPB];
    ss[threadIdx.x] = s; sq[threadIdx.x] = s2;
    __syncthreads();
    for (int o = TPB / 2; o > 0; o >>= 1) {
        if (threadIdx.x < o) {
            ss[threadIdx.x] += ss[threadIdx.x + o];
            sq[threadIdx.x] += sq[threadIdx.x + o];
        }
        __syncthreads();
    }
    if (threadIdx.x == 0) {
        float m = ss[0] / (float)per;
        stats[2 * c]     = m;
        stats[2 * c + 1] = sq[0] / (float)per - m * m;
    }
}

__global__ __launch_bounds__(TPB)
void bn_apply_kernel(const float* __restrict__ x, const float* __restrict__ stats,
                     float* __restrict__ y, int C, int HW, int total, int relu)
{
    int i = blockIdx.x * TPB + threadIdx.x;
    if (i >= total) return;
    int c = (i / HW) % C;
    float m = stats[2 * c], v = stats[2 * c + 1];
    float r = (x[i] - m) * rsqrtf(v + 1e-5f);
    y[i] = relu ? fmaxf(r, 0.f) : r;
}

// =====================================================================
// CBAM helpers
// =====================================================================
__global__ __launch_bounds__(TPB)
void chan_pool_kernel(const float* __restrict__ x, float* __restrict__ mean,
                      float* __restrict__ mx, int HW)
{
    int bc = blockIdx.x;
    const float* p = x + (size_t)bc * HW;
    float s = 0.f, m = -3.4e38f;
    for (int i = threadIdx.x; i < HW; i += TPB) {
        float v = p[i];
        s += v; m = fmaxf(m, v);
    }
    __shared__ float ss[TPB], sm[TPB];
    ss[threadIdx.x] = s; sm[threadIdx.x] = m;
    __syncthreads();
    for (int o = TPB / 2; o > 0; o >>= 1) {
        if (threadIdx.x < o) {
            ss[threadIdx.x] += ss[threadIdx.x + o];
            sm[threadIdx.x] = fmaxf(sm[threadIdx.x], sm[threadIdx.x + o]);
        }
        __syncthreads();
    }
    if (threadIdx.x == 0) { mean[bc] = ss[0] / (float)HW; mx[bc] = sm[0]; }
}

__global__ __launch_bounds__(64)
void cbam_mlp_kernel(const float* __restrict__ mean, const float* __restrict__ mx,
                     const float* __restrict__ w1, const float* __restrict__ w2,
                     float* __restrict__ ca, int C, int Cr)
{
    int b = blockIdx.x, c = threadIdx.x;
    __shared__ float vm[64], vx[64], hm[16], hx[16];
    vm[c] = mean[b * C + c];
    vx[c] = mx[b * C + c];
    __syncthreads();
    if (c < Cr) {
        float a = 0.f, d = 0.f;
        for (int i = 0; i < C; ++i) {
            float ww = w1[c * C + i];
            a += ww * vm[i];
            d += ww * vx[i];
        }
        hm[c] = fmaxf(a, 0.f);
        hx[c] = fmaxf(d, 0.f);
    }
    __syncthreads();
    float o = 0.f;
    for (int j = 0; j < Cr; ++j) o += w2[c * Cr + j] * (hm[j] + hx[j]);
    ca[b * C + c] = 1.f / (1.f + expf(-o));
}

__global__ __launch_bounds__(TPB)
void mul_ca_kernel(const float* __restrict__ x, const float* __restrict__ ca,
                   float* __restrict__ y, int HW, int total)
{
    int i = blockIdx.x * TPB + threadIdx.x;
    if (i >= total) return;
    y[i] = x[i] * ca[i / HW];
}

__global__ __launch_bounds__(TPB)
void spatial_pool_kernel(const float* __restrict__ x, float* __restrict__ sp,
                         int C, int HW, int total)
{
    int i = blockIdx.x * TPB + threadIdx.x;
    if (i >= total) return;               // total = B*HW
    int b = i / HW, r = i - (i / HW) * HW;
    const float* p = x + (size_t)b * C * HW + r;
    float s = 0.f, m = -3.4e38f;
    for (int c = 0; c < C; ++c) {
        float v = p[(size_t)c * HW];
        s += v; m = fmaxf(m, v);
    }
    sp[((size_t)b * 2 + 0) * HW + r] = s / (float)C;
    sp[((size_t)b * 2 + 1) * HW + r] = m;
}

__global__ __launch_bounds__(TPB)
void conv_sp7_kernel(const float* __restrict__ sp, const float* __restrict__ w,
                     float* __restrict__ out, int H, int W, int total)
{
    int i = blockIdx.x * TPB + threadIdx.x;
    if (i >= total) return;               // total = B*H*W
    int HW = H * W;
    int b = i / HW, r = i - b * HW;
    int h = r / W, x = r - h * W;
    float acc = 0.f;
    for (int ci = 0; ci < 2; ++ci)
        for (int kh = 0; kh < 7; ++kh) {
            int yy = h + kh - 3;
            if (yy < 0 || yy >= H) continue;
            for (int kw = 0; kw < 7; ++kw) {
                int xx = x + kw - 3;
                if (xx < 0 || xx >= W) continue;
                acc += w[(ci * 7 + kh) * 7 + kw] *
                       sp[((size_t)b * 2 + ci) * HW + yy * W + xx];
            }
        }
    out[i] = 1.f / (1.f + expf(-acc));
}

__global__ __launch_bounds__(TPB)
void mul_sp_kernel(const float* __restrict__ x, const float* __restrict__ s,
                   float* __restrict__ y, int C, int HW, int total)
{
    int i = blockIdx.x * TPB + threadIdx.x;
    if (i >= total) return;
    int b = i / (C * HW);
    int hw = i % HW;
    y[i] = x[i] * s[(size_t)b * HW + hw];
}

// =====================================================================
// Deformable-conv bilinear gather: t:(B,C,H,W), off:(B,18,H,W) ->
// xo:(B,C,3H,3W) where xo[b,c,3h+ki,3w+kj] = bilinear(t_padded, p(n=ki*3+kj))
// =====================================================================
__global__ __launch_bounds__(TPB)
void deform_gather_kernel(const float* __restrict__ t, const float* __restrict__ off,
                          float* __restrict__ xo, int B, int C, int H, int W)
{
    long total = (long)B * C * H * W * 9;
    long i = (long)blockIdx.x * TPB + threadIdx.x;
    if (i >= total) return;
    int nk = (int)(i % 9);  long r = i / 9;
    int w  = (int)(r % W);  r /= W;
    int h  = (int)(r % H);  r /= H;
    int c  = (int)(r % C);
    int b  = (int)(r / C);
    int ki = nk / 3, kj = nk % 3;
    int HW = H * W;
    const float* offb = off + (size_t)b * 18 * HW;
    float oh = offb[(size_t)(2 * nk)     * HW + h * W + w];
    float ow = offb[(size_t)(2 * nk + 1) * HW + h * W + w];
    float ph = (float)(h + ki) + oh;     // (h+1) + (ki-1) in padded coords
    float pw = (float)(w + kj) + ow;
    int Hp = H + 2, Wp = W + 2;
    if (ph < 1.f || ph > (float)(Hp - 2)) ph = floorf(ph);
    if (pw < 1.f || pw > (float)(Wp - 2)) pw = floorf(pw);
    ph = fminf(fmaxf(ph, 0.f), (float)(Hp - 1));
    pw = fminf(fmaxf(pw, 0.f), (float)(Wp - 1));
    int h0 = (int)floorf(ph), w0 = (int)floorf(pw);
    int h1 = min(h0 + 1, Hp - 1), w1 = min(w0 + 1, Wp - 1);
    float ah = ph - (float)h0, aw = pw - (float)w0;
    const float* tb = t + ((size_t)b * C + c) * HW;
    auto S = [&](int yy, int xx) -> float {
        yy -= 1; xx -= 1;
        return (yy >= 0 && yy < H && xx >= 0 && xx < W) ? tb[yy * W + xx] : 0.f;
    };
    float v = (1.f - ah) * (1.f - aw) * S(h0, w0) + (1.f - ah) * aw * S(h0, w1)
            + ah * (1.f - aw) * S(h1, w0) + ah * aw * S(h1, w1);
    xo[(((size_t)b * C + c) * (3 * H) + (h * 3 + ki)) * (3 * W) + (w * 3 + kj)] = v;
}

// =====================================================================
// Elementwise / concat helpers
// =====================================================================
__global__ __launch_bounds__(TPB)
void add2_kernel(const float* __restrict__ a, const float* __restrict__ b,
                 float* __restrict__ y, int n)
{
    int i = blockIdx.x * TPB + threadIdx.x;
    if (i < n) y[i] = a[i] + b[i];
}

__global__ __launch_bounds__(TPB)
void copy_ch_kernel(const float* __restrict__ src, float* __restrict__ dst,
                    int B, int Cs, int cs, int Cd, int cd, int Cc, int HW)
{
    int i = blockIdx.x * TPB + threadIdx.x;
    int total = B * Cc * HW;
    if (i >= total) return;
    int hw = i % HW;
    int t  = i / HW;
    int c  = t % Cc;
    int b  = t / Cc;
    dst[((size_t)b * Cd + cd + c) * HW + hw] = src[((size_t)b * Cs + cs + c) * HW + hw];
}

// =====================================================================
// Host-side launchers
// =====================================================================
static inline dim3 g1(size_t n) { return dim3((unsigned)((n + TPB - 1) / TPB)); }

static void conv_launch(hipStream_t s, const float* x, const float* x2, int Cin1,
                        const float* w, const float* bias, float* y,
                        int Nb, int Cin, int H, int W, int Cout,
                        int ks, int pad, int dil, int stride, int relu)
{
    int eff = dil * (ks - 1) + 1;
    int Ho = (H + 2 * pad - eff) / stride + 1;
    int Wo = (W + 2 * pad - eff) / stride + 1;
    int Npix = Nb * Ho * Wo;
    if (Cout > 32) {        // 64-channel outputs: 64x64 block tile
        dim3 g((Npix + 63) / 64, (Cout + 63) / 64);
        conv_wmma_kernel<4, 2><<<g, TPB, 0, s>>>(x, x2, Cin1, w, bias, y, Nb, Cin, H, W, Cout, Ho, Wo, ks, pad, dil, stride, relu);
    } else if (Cout > 16) { // 18-channel offset conv: 32x128 block tile
        dim3 g((Npix + 127) / 128, (Cout + 31) / 32);
        conv_wmma_kernel<2, 4><<<g, TPB, 0, s>>>(x, x2, Cin1, w, bias, y, Nb, Cin, H, W, Cout, Ho, Wo, ks, pad, dil, stride, relu);
    } else {                // 16-channel outputs: 16x256 block tile
        dim3 g((Npix + 255) / 256, (Cout + 15) / 16);
        conv_wmma_kernel<1, 8><<<g, TPB, 0, s>>>(x, x2, Cin1, w, bias, y, Nb, Cin, H, W, Cout, Ho, Wo, ks, pad, dil, stride, relu);
    }
}

extern "C" void kernel_launch(void* const* d_in, const int* in_sizes, int n_in,
                              void* d_out, int out_size, void* d_ws, size_t ws_size,
                              hipStream_t stream)
{
    (void)in_sizes; (void)n_in; (void)out_size; (void)ws_size;

    const float* lf        = (const float*)d_in[0];
    const float* hf        = (const float*)d_in[1];
    const float* w_conv1_1 = (const float*)d_in[2];
    const float* b_conv1_1 = (const float*)d_in[3];
    const float* w_conv    = (const float*)d_in[4];
    const float* w_offset  = (const float*)d_in[5];
    const float* b_offset  = (const float*)d_in[6];
    const float* w_deform  = (const float*)d_in[7];
    const float* w_dil[4]  = { (const float*)d_in[8], (const float*)d_in[9],
                               (const float*)d_in[10], (const float*)d_in[11] };
    const float* w_conv1   = (const float*)d_in[12];
    const float* w_conv2   = (const float*)d_in[13];
    const float* w_fc1     = (const float*)d_in[14];
    const float* w_fc2     = (const float*)d_in[15];
    const float* w_sp      = (const float*)d_in[16];
    const float* w_conv1_2 = (const float*)d_in[17];
    const float* b_conv1_2 = (const float*)d_in[18];
    const float* w_conv3_3 = (const float*)d_in[19];

    const int B = 4, C = 64, c4 = 16, H = 256, W = 256, HW = H * W;
    const size_t S16 = (size_t)B * c4 * HW;
    const size_t S64 = (size_t)B * C * HW;
    const size_t S18 = (size_t)B * 18 * HW;
    const size_t SXO = (size_t)B * c4 * 9 * HW;

    float* base = (float*)d_ws;
    size_t off = 0;
    auto alloc = [&](size_t n) { float* p = base + off; off += n; return p; };

    float* prev0  = alloc(S16);
    float* extraB = alloc(S16);
    float* hfp    = alloc(S16);
    float* bufX   = alloc(S16);
    float* bufT   = alloc(S16);
    float* bufOff = alloc(S18);
    float* xo     = alloc(SXO);
    float* bufD   = alloc(S16);
    float* bufD2  = alloc(S16);
    float* e64    = alloc(S64);
    float* e64b   = alloc(S64);
    float* e64c   = alloc(S64);
    float* x01    = alloc(S16);
    float* x12    = alloc(S16);
    float* x23    = alloc(S16);
    float* x34    = alloc(S16);
    float* cat64  = alloc(S64);
    float* xx     = alloc(S64);
    float* sum    = alloc(S64);
    float* tmpO   = alloc(S64);
    float* spbuf  = alloc((size_t)B * 2 * HW);
    float* spsig  = alloc((size_t)B * HW);
    float* pm     = alloc((size_t)B * C);
    float* px     = alloc((size_t)B * C);
    float* ca     = alloc((size_t)B * C);
    float* stats  = alloc(2 * C);

    auto run_bn = [&](float* buf, int Cc, int relu) {
        bn_stats_kernel<<<Cc, TPB, 0, stream>>>(buf, stats, B, Cc, HW);
        size_t total = (size_t)B * Cc * HW;
        bn_apply_kernel<<<g1(total), TPB, 0, stream>>>(buf, stats, buf, Cc, HW, (int)total, relu);
    };

    // ---- hf path: relu(bn(conv1x1(hf) + bias)) -> 16 channels ----
    conv_launch(stream, hf, hf, C, w_conv1_1, b_conv1_1, hfp, B, C, H, W, c4, 1, 0, 1, 1, 0);
    run_bn(hfp, c4, 1);

    // ---- branch lambda ----
    auto branch = [&](const float* prev, const float* extra, const float* wd, int dil, float* outb) {
        // conv over fused concat([prev(16ch), hfp(16ch)]) -> 16 ch
        conv_launch(stream, prev, hfp, c4, w_conv, nullptr, bufX, B, 2 * c4, H, W, c4, 3, 1, 1, 1, 0);
        run_bn(bufX, c4, 0);
        add2_kernel<<<g1(S16), TPB, 0, stream>>>(bufX, extra, bufT, (int)S16);
        // deform: offsets then gather then stride-3 conv
        conv_launch(stream, bufT, bufT, c4, w_offset, b_offset, bufOff, B, c4, H, W, 18, 3, 1, 1, 1, 0);
        deform_gather_kernel<<<g1((size_t)B * c4 * HW * 9), TPB, 0, stream>>>(bufT, bufOff, xo, B, c4, H, W);
        conv_launch(stream, xo, xo, c4, w_deform, nullptr, bufD, B, c4, 3 * H, 3 * W, c4, 3, 0, 1, 3, 0);
        // dilated conv + bn
        conv_launch(stream, bufD, bufD, c4, wd, nullptr, bufD2, B, c4, H, W, c4, 3, dil, dil, 1, 0);
        run_bn(bufD2, c4, 0);
        // 16 -> 64 conv + bn
        conv_launch(stream, bufD2, bufD2, c4, w_conv1, nullptr, e64, B, c4, H, W, C, 3, 1, 1, 1, 0);
        run_bn(e64, C, 0);
        // CBAM
        chan_pool_kernel<<<B * C, TPB, 0, stream>>>(e64, pm, px, HW);
        cbam_mlp_kernel<<<B, 64, 0, stream>>>(pm, px, w_fc1, w_fc2, ca, C, 4);
        mul_ca_kernel<<<g1(S64), TPB, 0, stream>>>(e64, ca, e64b, HW, (int)S64);
        spatial_pool_kernel<<<g1((size_t)B * HW), TPB, 0, stream>>>(e64b, spbuf, C, HW, B * HW);
        conv_sp7_kernel<<<g1((size_t)B * HW), TPB, 0, stream>>>(spbuf, w_sp, spsig, H, W, B * HW);
        mul_sp_kernel<<<g1(S64), TPB, 0, stream>>>(e64b, spsig, e64c, C, HW, (int)S64);
        // 64 -> 16 conv + bn
        conv_launch(stream, e64c, e64c, C, w_conv2, nullptr, outb, B, C, H, W, c4, 3, 1, 1, 1, 0);
        run_bn(outb, c4, 0);
    };

    // ---- branch 1: prev = xc[0], extra = xc[1] ----
    copy_ch_kernel<<<g1(S16), TPB, 0, stream>>>(lf, prev0,  B, C, 0,  c4, 0, c4, HW);
    copy_ch_kernel<<<g1(S16), TPB, 0, stream>>>(lf, extraB, B, C, 16, c4, 0, c4, HW);
    branch(prev0, extraB, w_dil[0], 1, x01);

    // ---- branch 2: prev = x01, extra = x01 + xc[2] ----
    copy_ch_kernel<<<g1(S16), TPB, 0, stream>>>(lf, extraB, B, C, 32, c4, 0, c4, HW);
    add2_kernel<<<g1(S16), TPB, 0, stream>>>(extraB, x01, extraB, (int)S16);
    branch(x01, extraB, w_dil[1], 2, x12);

    // ---- branch 3: prev = x12, extra = x12 + xc[3] ----
    copy_ch_kernel<<<g1(S16), TPB, 0, stream>>>(lf, extraB, B, C, 48, c4, 0, c4, HW);
    add2_kernel<<<g1(S16), TPB, 0, stream>>>(extraB, x12, extraB, (int)S16);
    branch(x12, extraB, w_dil[2], 3, x23);

    // ---- branch 4: prev = x23, extra = x23 ----
    branch(x23, x23, w_dil[3], 4, x34);

    // ---- merge: xx = relu(bn(conv1x1(cat(x01..x34)) + bias)) ----
    copy_ch_kernel<<<g1(S16), TPB, 0, stream>>>(x01, cat64, B, c4, 0, C, 0,  c4, HW);
    copy_ch_kernel<<<g1(S16), TPB, 0, stream>>>(x12, cat64, B, c4, 0, C, 16, c4, HW);
    copy_ch_kernel<<<g1(S16), TPB, 0, stream>>>(x23, cat64, B, c4, 0, C, 32, c4, HW);
    copy_ch_kernel<<<g1(S16), TPB, 0, stream>>>(x34, cat64, B, c4, 0, C, 48, c4, HW);
    conv_launch(stream, cat64, cat64, C, w_conv1_2, b_conv1_2, xx, B, C, H, W, C, 1, 0, 1, 1, 0);
    run_bn(xx, C, 1);

    // ---- out = bn(conv3x3(lf + xx)) ----
    add2_kernel<<<g1(S64), TPB, 0, stream>>>(lf, xx, sum, (int)S64);
    conv_launch(stream, sum, sum, C, w_conv3_3, nullptr, tmpO, B, C, H, W, C, 3, 1, 1, 1, 0);
    bn_stats_kernel<<<C, TPB, 0, stream>>>(tmpO, stats, B, C, HW);
    bn_apply_kernel<<<g1(S64), TPB, 0, stream>>>(tmpO, stats, (float*)d_out, C, HW, (int)S64, 0);
}
